// DeformablePatchEmbedding_64742337020513
// MI455X (gfx1250) — compile-verified, matrix operand order checked by reading, not run on hardware
//
#include <hip/hip_runtime.h>
#include <hip/hip_bf16.h>

// ---------------------------------------------------------------------------
// Deformable patch embedding, fused for MI455X (gfx1250, wave32, WMMA).
//   B=64, CIN=3, H=W=384, K=S=16 -> HO=WO=24, 36864 patches, EMB=768.
//   Per workgroup: 32 patches (two 16-row WMMA M-tiles -> each B fragment
//   feeds two v_wmma ops, halving L2 weight traffic per FLOP).
//   Stage A: offsets = patches(32x768) x offset_wT(768x512)   [WMMA bf16]
//   Stage B: bilinear gather (image is L2-resident: 113MB < 192MB L2)
//   Stage C: out = sampled(32x768) x proj_wT(768x768) + bias  [WMMA bf16]
// ---------------------------------------------------------------------------

typedef __attribute__((ext_vector_type(16))) __bf16 v16bf;
typedef __attribute__((ext_vector_type(8)))  float  v8f;

#define N_B     64
#define N_H     384
#define N_W     384
#define N_WO    24
#define PPI     576          // patches per image (24*24)
#define EMB     768
#define PATCH   768          // CIN*K*K
#define OFFC    512          // 2*K*K
#define NPATCH  36864        // N_B * PPI
#define MTILE   32           // patches per workgroup

#define A_STRIDE   776       // ushort stride, 32x768 LDS A (pad vs bank conflicts)
#define OFF_STRIDE 516       // float stride, 32x512 LDS offsets

struct alignas(32) Frag { uint4 lo; uint4 hi; };

__device__ __forceinline__ unsigned short f2bf(float f) {
    unsigned int u = __float_as_uint(f);
    u += 0x7FFFu + ((u >> 16) & 1u);   // round-to-nearest-even
    return (unsigned short)(u >> 16);
}

__global__ void convert_weights(const float* __restrict__ ow,   // [512*768]
                                const float* __restrict__ pw,   // [768*768]
                                unsigned short* __restrict__ owb,
                                unsigned short* __restrict__ pwb) {
    const int n1 = OFFC * PATCH;
    const int n2 = EMB * PATCH;
    for (int i = blockIdx.x * blockDim.x + threadIdx.x; i < n1 + n2;
         i += gridDim.x * blockDim.x) {
        if (i < n1) owb[i] = f2bf(ow[i]);
        else        pwb[i - n1] = f2bf(pw[i - n1]);
    }
}

__launch_bounds__(256)
__global__ void deform_patch_embed(const float* __restrict__ pix,          // [64,3,384,384]
                                   const unsigned short* __restrict__ owb, // bf16 [512][768]
                                   const float* __restrict__ ob,           // [512]
                                   const unsigned short* __restrict__ pwb, // bf16 [768][768]
                                   const float* __restrict__ pb,           // [768]
                                   float* __restrict__ out)                // [36864][768]
{
    __shared__ unsigned short sA[MTILE * A_STRIDE];     // bf16 32x768 A (reused)
    __shared__ float          sOff[MTILE * OFF_STRIDE]; // f32 32x512 offsets

    const int tid   = threadIdx.x;
    const int wave  = tid >> 5;
    const int lane  = tid & 31;
    const int pbase = blockIdx.x * MTILE;

    const int col = lane & 15;          // A row (within M-tile) / B col / C col
    const int hi  = lane >> 4;          // half-wave select
    const int kb  = hi * 8;             // K sub-chunk base within fragment

    // -------- Stage 1: patch pixels -> bf16 A in LDS (float4 loads) --------
    // 32 rows x 192 float4 = 6144 float4; 24 iters x 256 threads.
    for (int it = 0; it < 24; ++it) {
        int idx = it * 256 + tid;
        int m   = idx / 192;
        int k   = (idx - m * 192) * 4;          // multiple of 4 -> same row i
        int p   = pbase + m;
        int b   = p / PPI;
        int rem = p - b * PPI;
        int ho  = rem / N_WO;
        int wo  = rem - ho * N_WO;
        int c   = k >> 8;
        int q   = k & 255;
        int i   = q >> 4;
        int j   = q & 15;                        // 0,4,8,12
        const float4 v = *(const float4*)
            (pix + (((size_t)(b * 3 + c) * N_H) + ho * 16 + i) * N_W + wo * 16 + j);
        uint2 packed;
        packed.x = (unsigned)f2bf(v.x) | ((unsigned)f2bf(v.y) << 16);
        packed.y = (unsigned)f2bf(v.z) | ((unsigned)f2bf(v.w) << 16);
        *(uint2*)(&sA[m * A_STRIDE + k]) = packed;
    }
    __syncthreads();

    // -------- Stage 2: offset GEMM (32x768)x(768x512), 2 M-tiles/wave -------
    for (int nt = 0; nt < 4; ++nt) {
        const int n0 = (wave * 4 + nt) * 16;
        v8f acc0 = {};
        v8f acc1 = {};
        const unsigned short* brow = owb + (size_t)(n0 + col) * PATCH;
        for (int kt = 0; kt < 24; ++kt) {
            const int k0 = kt * 32;
            Frag fb, fa0, fa1;
            fb.lo  = *(const uint4*)(&brow[k0 + kb]);
            fb.hi  = *(const uint4*)(&brow[k0 + 16 + kb]);
            fa0.lo = *(const uint4*)(&sA[col * A_STRIDE + k0 + kb]);
            fa0.hi = *(const uint4*)(&sA[col * A_STRIDE + k0 + 16 + kb]);
            fa1.lo = *(const uint4*)(&sA[(col + 16) * A_STRIDE + k0 + kb]);
            fa1.hi = *(const uint4*)(&sA[(col + 16) * A_STRIDE + k0 + 16 + kb]);
            v16bf vb  = __builtin_bit_cast(v16bf, fb);
            v16bf va0 = __builtin_bit_cast(v16bf, fa0);
            v16bf va1 = __builtin_bit_cast(v16bf, fa1);
            acc0 = __builtin_amdgcn_wmma_f32_16x16x32_bf16(
                       false, va0, false, vb, (short)0, acc0, false, false);
            acc1 = __builtin_amdgcn_wmma_f32_16x16x32_bf16(
                       false, va1, false, vb, (short)0, acc1, false, false);
        }
        const float bias = ob[n0 + col];
#pragma unroll
        for (int r = 0; r < 8; ++r) {
            sOff[(r + 8 * hi) * OFF_STRIDE + n0 + col]        = acc0[r] + bias;
            sOff[(16 + r + 8 * hi) * OFF_STRIDE + n0 + col]   = acc1[r] + bias;
        }
    }
    __syncthreads();

    // -------- Stage 3: deformable bilinear gather -> bf16 A in LDS ----------
    // 32 rows x 768 = 24576 samples; 96 iters x 256 threads.
    for (int it = 0; it < 96; ++it) {
        int idx = it * 256 + tid;
        int m   = idx / PATCH;
        int k   = idx - m * PATCH;
        int p   = pbase + m;
        int b   = p / PPI;
        int rem = p - b * PPI;
        int ho  = rem / N_WO;
        int wo  = rem - ho * N_WO;
        int c   = k >> 8;
        int q   = k & 255;
        int i   = q >> 4;
        int j   = q & 15;

        float dy = sOff[m * OFF_STRIDE + 2 * q];
        float dx = sOff[m * OFF_STRIDE + 2 * q + 1];
        float py = (float)(ho * 16 + i) + dy;
        float px = (float)(wo * 16 + j) + dx;

        float y0f = floorf(py), x0f = floorf(px);
        float wy1 = py - y0f, wx1 = px - x0f;
        float wy0 = 1.f - wy1, wx0 = 1.f - wx1;
        int y0 = (int)y0f, x0 = (int)x0f;

        const float* img = pix + (size_t)(b * 3 + c) * (N_H * N_W);
        float v = 0.f;
        if ((unsigned)y0 < (unsigned)N_H && (unsigned)x0 < (unsigned)N_W)
            v += img[y0 * N_W + x0] * (wy0 * wx0);
        if ((unsigned)y0 < (unsigned)N_H && (unsigned)(x0 + 1) < (unsigned)N_W)
            v += img[y0 * N_W + x0 + 1] * (wy0 * wx1);
        if ((unsigned)(y0 + 1) < (unsigned)N_H && (unsigned)x0 < (unsigned)N_W)
            v += img[(y0 + 1) * N_W + x0] * (wy1 * wx0);
        if ((unsigned)(y0 + 1) < (unsigned)N_H && (unsigned)(x0 + 1) < (unsigned)N_W)
            v += img[(y0 + 1) * N_W + x0 + 1] * (wy1 * wx1);

        sA[m * A_STRIDE + k] = f2bf(v);
    }
    __syncthreads();

    // -------- Stage 4: projection GEMM (32x768)x(768x768) + bias ------------
    for (int nt = 0; nt < 6; ++nt) {
        const int n0 = (wave * 6 + nt) * 16;
        v8f acc0 = {};
        v8f acc1 = {};
        const unsigned short* brow = pwb + (size_t)(n0 + col) * PATCH;
        for (int kt = 0; kt < 24; ++kt) {
            const int k0 = kt * 32;
            Frag fb, fa0, fa1;
            fb.lo  = *(const uint4*)(&brow[k0 + kb]);
            fb.hi  = *(const uint4*)(&brow[k0 + 16 + kb]);
            fa0.lo = *(const uint4*)(&sA[col * A_STRIDE + k0 + kb]);
            fa0.hi = *(const uint4*)(&sA[col * A_STRIDE + k0 + 16 + kb]);
            fa1.lo = *(const uint4*)(&sA[(col + 16) * A_STRIDE + k0 + kb]);
            fa1.hi = *(const uint4*)(&sA[(col + 16) * A_STRIDE + k0 + 16 + kb]);
            v16bf vb  = __builtin_bit_cast(v16bf, fb);
            v16bf va0 = __builtin_bit_cast(v16bf, fa0);
            v16bf va1 = __builtin_bit_cast(v16bf, fa1);
            acc0 = __builtin_amdgcn_wmma_f32_16x16x32_bf16(
                       false, va0, false, vb, (short)0, acc0, false, false);
            acc1 = __builtin_amdgcn_wmma_f32_16x16x32_bf16(
                       false, va1, false, vb, (short)0, acc1, false, false);
        }
        const float bias = pb[n0 + col];
#pragma unroll
        for (int r = 0; r < 8; ++r) {
            int row0 = r + 8 * hi;
            out[(size_t)(pbase + row0) * EMB + n0 + col]      = acc0[r] + bias;
            out[(size_t)(pbase + 16 + row0) * EMB + n0 + col] = acc1[r] + bias;
        }
    }
}

extern "C" void kernel_launch(void* const* d_in, const int* in_sizes, int n_in,
                              void* d_out, int out_size, void* d_ws, size_t ws_size,
                              hipStream_t stream) {
    const float* pix = (const float*)d_in[0];  // [64,3,384,384]
    const float* ow  = (const float*)d_in[1];  // [512,3,16,16]
    const float* ob  = (const float*)d_in[2];  // [512]
    const float* pw  = (const float*)d_in[3];  // [768,3,16,16]
    const float* pb  = (const float*)d_in[4];  // [768]

    unsigned short* owb = (unsigned short*)d_ws;            // bf16 [512*768]
    unsigned short* pwb = owb + (size_t)OFFC * PATCH;       // bf16 [768*768]

    convert_weights<<<256, 256, 0, stream>>>(ow, pw, owb, pwb);
    deform_patch_embed<<<NPATCH / MTILE, 256, 0, stream>>>(
        pix, owb, ob, pwb, pb, (float*)d_out);
}
